// MambaBlock_36790689857878
// MI455X (gfx1250) — matrix operand order repaired and checked by reading, not compiled, not run on hardware
//
#include <hip/hip_runtime.h>
#include <hip/hip_bf16.h>
#include <math.h>

// ---------------- problem constants (match reference) ----------------
#define BQ      2
#define LQ      1024
#define D_MODEL 1024
#define D_INNER 2048
#define D_STATE 16
#define DT_RANK 64
#define D_CONV  4
#define N_XP    (DT_RANK + 2 * D_STATE)   // 96
#define N_XPP   128                       // x_dbl padded stride (96 -> 128)
#define ROWS    (BQ * LQ)                 // 2048

// ---------------- WMMA vector types ----------------
typedef __attribute__((ext_vector_type(16))) __bf16 v16bf;
typedef __attribute__((ext_vector_type(8)))  __bf16 v8bf;
typedef __attribute__((ext_vector_type(8)))  float  v8f;

// fp32 -> bf16, round-to-nearest-even (prep, conv and scan kernels)
static __device__ __forceinline__ __bf16 f2bf(float f) {
    union { float f; unsigned u; } in; in.f = f;
    unsigned u = in.u;
    unsigned r = u + 0x7FFFu + ((u >> 16) & 1u);
    unsigned short h = (unsigned short)(r >> 16);
    __bf16 out;
    __builtin_memcpy(&out, &h, 2);
    return out;
}

// =====================================================================
// elementwise fp32 -> bf16
// =====================================================================
__global__ __launch_bounds__(256) void cvt_bf16_kernel(
    const float* __restrict__ in, __bf16* __restrict__ out, int n)
{
    int i = blockIdx.x * blockDim.x + threadIdx.x;
    if (i < n) out[i] = f2bf(in[i]);
}

// fp32 (N,K) -> bf16 (Npad,K), rows >= N zero-filled
__global__ __launch_bounds__(256) void cvt_pad_bf16_kernel(
    const float* __restrict__ in, __bf16* __restrict__ out,
    int N, int K, int Npad)
{
    int i = blockIdx.x * blockDim.x + threadIdx.x;
    if (i >= Npad * K) return;
    int row = i / K;
    out[i] = (row < N) ? f2bf(in[i]) : (__bf16)0.0f;
}

// =====================================================================
// GEMM: C[M,N] = act( A[M,K](bf16, stride lda) @ W[N,K]^T + bias )
// Wave tile 32(M) x 64(N): 2 A frags x 4 B frags -> 8 WMMA per K-step.
// Requires M%32==0, N%64==0 (guaranteed by padding) -> branch-free loop,
// EXEC all ones at every WMMA.  K is compile-time; loop unrolled x4 so
// the scheduler can hoist the next group's loads above current WMMAs.
// ACT: 0 = none, 2 = softplus
// =====================================================================
template <int ACT, int K>
__global__ __launch_bounds__(128) void wmma_gemm_kernel(
    const __bf16* __restrict__ A, int lda,
    const __bf16* __restrict__ W,     // (N, K) row-major bf16
    const float*  __restrict__ bias,  // (N) or nullptr
    float* __restrict__ C,            // (M, N) row-major f32
    int M, int N)
{
    const int lane   = threadIdx.x & 31;
    const int wid    = threadIdx.x >> 5;
    const int tilesN = N >> 6;
    const int gw     = blockIdx.x * 4 + wid;
    if (gw >= (M >> 5) * tilesN) return;          // whole-wave exit

    const int tm = gw / tilesN;
    const int tn = gw % tilesN;

    const int r  = lane & 15;          // row-in-subtile (A) / col (B)
    const int kh = (lane >> 4) << 4;   // A half-select: +0 or +16
    const int kb = (lane >> 4) << 4;   // B half-select: +0 or +16

    const __bf16* pa0 = A + (size_t)(tm * 32 + r) * (size_t)lda + kh;
    const __bf16* pa1 = pa0 + (size_t)16 * lda;
    const __bf16* pw0 = W + (size_t)(tn * 64 + r) * (size_t)K + kb;
    const __bf16* pw1 = pw0 + (size_t)16 * K;
    const __bf16* pw2 = pw0 + (size_t)32 * K;
    const __bf16* pw3 = pw0 + (size_t)48 * K;

    v8f acc[2][4];
#pragma unroll
    for (int m = 0; m < 2; ++m)
#pragma unroll
        for (int j = 0; j < 4; ++j) acc[m][j] = v8f{};

#pragma unroll 4
    for (int kk = 0; kk < K; kk += 32) {
        union { v16bf v; v8bf h[2]; } a0, a1;
        a0.h[0] = *(const v8bf*)(pa0 + kk);
        a0.h[1] = *(const v8bf*)(pa0 + kk + 8);
        a1.h[0] = *(const v8bf*)(pa1 + kk);
        a1.h[1] = *(const v8bf*)(pa1 + kk + 8);
        const v16bf b0 = *(const v16bf*)(pw0 + kk);
        const v16bf b1 = *(const v16bf*)(pw1 + kk);
        const v16bf b2 = *(const v16bf*)(pw2 + kk);
        const v16bf b3 = *(const v16bf*)(pw3 + kk);

        acc[0][0] = __builtin_amdgcn_wmma_f32_16x16x32_bf16(false, a0.v, false, b0, (short)0, acc[0][0], false, false);
        acc[1][0] = __builtin_amdgcn_wmma_f32_16x16x32_bf16(false, a1.v, false, b0, (short)0, acc[1][0], false, false);
        acc[0][1] = __builtin_amdgcn_wmma_f32_16x16x32_bf16(false, a0.v, false, b1, (short)0, acc[0][1], false, false);
        acc[1][1] = __builtin_amdgcn_wmma_f32_16x16x32_bf16(false, a1.v, false, b1, (short)0, acc[1][1], false, false);
        acc[0][2] = __builtin_amdgcn_wmma_f32_16x16x32_bf16(false, a0.v, false, b2, (short)0, acc[0][2], false, false);
        acc[1][2] = __builtin_amdgcn_wmma_f32_16x16x32_bf16(false, a1.v, false, b2, (short)0, acc[1][2], false, false);
        acc[0][3] = __builtin_amdgcn_wmma_f32_16x16x32_bf16(false, a0.v, false, b3, (short)0, acc[0][3], false, false);
        acc[1][3] = __builtin_amdgcn_wmma_f32_16x16x32_bf16(false, a1.v, false, b3, (short)0, acc[1][3], false, false);
    }

    // store D: VGPR i of lane l -> row (i + 8*(l>=16)), col (l&15)
    const int rbase = tm * 32 + ((lane >> 4) << 3);
    const int ncol  = tn * 64 + (lane & 15);
#pragma unroll
    for (int j = 0; j < 4; ++j) {
        const int n = ncol + j * 16;
        const float bv = bias ? bias[n] : 0.0f;
#pragma unroll
        for (int m = 0; m < 2; ++m) {
#pragma unroll
            for (int i = 0; i < 8; ++i) {
                float v = acc[m][j][i] + bv;
                if (ACT == 2) v = (v > 20.0f) ? v : log1pf(expf(v)); // softplus
                C[(size_t)(rbase + 16 * m + i) * (size_t)N + n] = v;
            }
        }
    }
}

// =====================================================================
// Causal depthwise conv1d (k=4, left pad 3) + SiLU on the x branch.
// Writes f32 (for scan) and bf16 (for x_proj GEMM).
// =====================================================================
__global__ __launch_bounds__(256) void conv_silu_kernel(
    const float* __restrict__ xz,       // (B*L, 2*D_INNER)
    const float* __restrict__ conv_w,   // (D_INNER, 1, 4)
    const float* __restrict__ conv_b,   // (D_INNER)
    float*  __restrict__ xact,          // (B*L, D_INNER) f32
    __bf16* __restrict__ xact_bf)       // (B*L, D_INNER) bf16
{
    const int idx = blockIdx.x * blockDim.x + threadIdx.x;
    if (idx >= ROWS * D_INNER) return;
    const int d  = idx % D_INNER;
    const int bl = idx / D_INNER;
    const int l  = bl % LQ;
    const int b  = bl / LQ;

    float acc = conv_b[d];
#pragma unroll
    for (int j = 0; j < D_CONV; ++j) {
        const int ls = l - (D_CONV - 1) + j;
        if (ls >= 0)
            acc += conv_w[d * D_CONV + j] *
                   xz[((size_t)(b * LQ + ls)) * (2 * D_INNER) + d];
    }
    const float s = acc / (1.0f + expf(-acc));   // silu
    xact[idx]    = s;
    xact_bf[idx] = f2bf(s);
}

// =====================================================================
// Selective scan fused with D-skip and z-gate; bf16 output for out_proj.
// One thread per (b, d) channel, serial over L.  B_t/C_t loads are
// wave-uniform broadcasts; dt/x/z loads coalesce over d.
// =====================================================================
__global__ __launch_bounds__(256) void scan_kernel(
    const float* __restrict__ xz,     // (B*L, 2*D_INNER)  (z branch)
    const float* __restrict__ xact,   // (B*L, D_INNER)
    const float* __restrict__ xdbl,   // (B*L, N_XPP): [dt_lr | B | C | pad]
    const float* __restrict__ dtb,    // (B*L, D_INNER) softplus(dt)
    const float* __restrict__ A_log,  // (D_INNER, D_STATE)
    const float* __restrict__ Dvec,   // (D_INNER)
    __bf16* __restrict__ yfin_bf)     // (B*L, D_INNER) bf16
{
    const int t = blockIdx.x * blockDim.x + threadIdx.x;
    if (t >= BQ * D_INNER) return;
    const int d = t % D_INNER;
    const int b = t / D_INNER;

    float Ad[D_STATE];
#pragma unroll
    for (int s = 0; s < D_STATE; ++s) Ad[s] = -expf(A_log[d * D_STATE + s]);
    const float Dd = Dvec[d];

    float h[D_STATE];
#pragma unroll
    for (int s = 0; s < D_STATE; ++s) h[s] = 0.0f;

    for (int l = 0; l < LQ; ++l) {
        const size_t rb = (size_t)(b * LQ + l);
        const float dt = dtb[rb * D_INNER + d];
        const float xt = xact[rb * D_INNER + d];
        const float* bc = xdbl + rb * N_XPP;

        float y = 0.0f;
#pragma unroll
        for (int s = 0; s < D_STATE; ++s) {
            const float dA = expf(dt * Ad[s]);
            h[s] = dA * h[s] + dt * bc[DT_RANK + s] * xt;
            y += h[s] * bc[DT_RANK + D_STATE + s];
        }
        const float z  = xz[rb * (2 * D_INNER) + D_INNER + d];
        const float zs = z / (1.0f + expf(-z));
        yfin_bf[rb * D_INNER + d] = f2bf((y + xt * Dd) * zs);
    }
}

// =====================================================================
// launch
// =====================================================================
extern "C" void kernel_launch(void* const* d_in, const int* in_sizes, int n_in,
                              void* d_out, int out_size, void* d_ws, size_t ws_size,
                              hipStream_t stream) {
    (void)in_sizes; (void)n_in; (void)out_size; (void)ws_size;

    const float* x         = (const float*)d_in[0];
    const float* in_proj_w = (const float*)d_in[1];
    const float* in_proj_b = (const float*)d_in[2];
    const float* conv_w    = (const float*)d_in[3];
    const float* conv_b    = (const float*)d_in[4];
    const float* x_proj_w  = (const float*)d_in[5];
    const float* dt_proj_w = (const float*)d_in[6];
    const float* dt_proj_b = (const float*)d_in[7];
    const float* A_log     = (const float*)d_in[8];
    const float* Dvec      = (const float*)d_in[9];
    const float* out_proj_w= (const float*)d_in[10];
    const float* out_proj_b= (const float*)d_in[11];
    float* out = (float*)d_out;                       // (B, L, D_MODEL) f32

    // -------- workspace layout --------
    float* wsf  = (float*)d_ws;
    float* xz   = wsf;                                  // ROWS*4096
    float* xact = xz   + (size_t)ROWS * (2 * D_INNER);  // ROWS*2048
    float* xdbl = xact + (size_t)ROWS * D_INNER;        // ROWS*128 (padded)
    float* dtb  = xdbl + (size_t)ROWS * N_XPP;          // ROWS*2048
    __bf16* wsb     = (__bf16*)(dtb + (size_t)ROWS * D_INNER);
    __bf16* x_bf    = wsb;                                       // ROWS*1024
    __bf16* xact_bf = x_bf    + (size_t)ROWS * D_MODEL;          // ROWS*2048
    __bf16* xdbl_bf = xact_bf + (size_t)ROWS * D_INNER;          // ROWS*128
    __bf16* yfin_bf = xdbl_bf + (size_t)ROWS * N_XPP;            // ROWS*2048
    __bf16* w_in_bf = yfin_bf + (size_t)ROWS * D_INNER;          // 4096*1024
    __bf16* w_xp_bf = w_in_bf + (size_t)(2 * D_INNER) * D_MODEL; // 128*2048 (padded)
    __bf16* w_dt_bf = w_xp_bf + (size_t)N_XPP * D_INNER;         // 2048*64
    __bf16* w_out_bf= w_dt_bf + (size_t)D_INNER * DT_RANK;       // 1024*2048

    auto cvt = [&](const float* src, __bf16* dst, int n) {
        cvt_bf16_kernel<<<(n + 255) / 256, 256, 0, stream>>>(src, dst, n);
    };
    auto gemm_blocks = [](int M, int N) {
        return ((M / 32) * (N / 64) + 3) / 4;
    };

    // 0) bf16 operand prep
    cvt(x,          x_bf,     ROWS * D_MODEL);
    cvt(in_proj_w,  w_in_bf,  2 * D_INNER * D_MODEL);
    cvt(dt_proj_w,  w_dt_bf,  D_INNER * DT_RANK);
    cvt(out_proj_w, w_out_bf, D_MODEL * D_INNER);
    cvt_pad_bf16_kernel<<<(N_XPP * D_INNER + 255) / 256, 256, 0, stream>>>(
        x_proj_w, w_xp_bf, N_XP, D_INNER, N_XPP);

    // 1) in_proj: xz = x @ in_proj_w^T + b   (M=2048, N=4096, K=1024)
    wmma_gemm_kernel<0, D_MODEL><<<gemm_blocks(ROWS, 2 * D_INNER), 128, 0, stream>>>(
        x_bf, D_MODEL, w_in_bf, in_proj_b, xz, ROWS, 2 * D_INNER);

    // 2) causal depthwise conv + silu
    conv_silu_kernel<<<(ROWS * D_INNER + 255) / 256, 256, 0, stream>>>(
        xz, conv_w, conv_b, xact, xact_bf);

    // 3) x_proj: x_dbl = xact @ x_proj_w^T  (M=2048, N=128 padded, K=2048)
    wmma_gemm_kernel<0, D_INNER><<<gemm_blocks(ROWS, N_XPP), 128, 0, stream>>>(
        xact_bf, D_INNER, w_xp_bf, nullptr, xdbl, ROWS, N_XPP);

    // 3b) bf16 copy of x_dbl (dt GEMM input, lda = 128)
    cvt(xdbl, xdbl_bf, ROWS * N_XPP);

    // 4) dt = softplus(x_dbl[:, :64] @ dt_proj_w^T + b) (M=2048, N=2048, K=64)
    wmma_gemm_kernel<2, DT_RANK><<<gemm_blocks(ROWS, D_INNER), 128, 0, stream>>>(
        xdbl_bf, N_XPP, w_dt_bf, dt_proj_b, dtb, ROWS, D_INNER);

    // 5) selective scan + D-skip + z-gate -> bf16 y
    scan_kernel<<<(BQ * D_INNER + 255) / 256, 256, 0, stream>>>(
        xz, xact, xdbl, dtb, A_log, Dvec, yfin_bf);

    // 6) out_proj: out = y @ out_proj_w^T + b  (M=2048, N=1024, K=2048)
    wmma_gemm_kernel<0, D_INNER><<<gemm_blocks(ROWS, D_MODEL), 128, 0, stream>>>(
        yfin_bf, D_INNER, w_out_bf, out_proj_b, out, ROWS, D_MODEL);
}
